// Decoder_13494787244451
// MI455X (gfx1250) — matrix operand order repaired and checked by reading, not compiled, not run on hardware
//
#include <hip/hip_runtime.h>
#include <hip/hip_bf16.h>

// ---------------- constants ----------------
#define VOCAB 32000
#define EDIM  256
#define HDIM  1024
#define BATCH 32
#define TSTEP 32

typedef __attribute__((ext_vector_type(16))) __bf16 v16bf;
typedef __attribute__((ext_vector_type(8)))  float  v8f;

union ABFrag { v16bf bf; unsigned u[8]; };

// ---------------- helpers ----------------
__device__ __forceinline__ unsigned f32_to_bf16_bits(float f) {
    unsigned u = __float_as_uint(f);
    return (u + 0x7fffu + ((u >> 16) & 1u)) >> 16;  // round-to-nearest-even
}

// A fragment: 16x32 bf16, rows striped over lanes (lane%16=M), K over VGPRs:
// lanes 0-15 hold K=0..15, lanes 16-31 hold K=16..31 (matches load_b).
__device__ __forceinline__ void load_a(ABFrag& a, const unsigned short* src,
                                       int row0, int ld, int k0, int lane) {
    const int half = lane >> 4, m = lane & 15;
    const unsigned short* p = src + (size_t)(row0 + m) * ld + k0 + half * 16;
#pragma unroll
    for (int v = 0; v < 8; ++v) {
        const int k = 2 * v;
        a.u[v] = *(const unsigned*)(p + k);
    }
}

// B fragment: 32x16 bf16 (KxN), N = lane%16, lanes 0-15 hold K=0..15,
// lanes 16-31 hold K=16..31, two K per VGPR. Source W is (N-major rows, K contiguous):
// B[kk][nn] = W[(ncol0+nn)*ld + k0+kk]  (i.e. computing X @ W^T).
__device__ __forceinline__ void load_b(ABFrag& b, const unsigned short* W,
                                       int ncol0, int ld, int k0, int lane) {
    const int kh = (lane >> 4) * 16, n = lane & 15;
    const unsigned short* p = W + (size_t)(ncol0 + n) * ld + k0 + kh;
#pragma unroll
    for (int v = 0; v < 8; ++v)
        b.u[v] = *(const unsigned*)(p + 2 * v);
}

__device__ __forceinline__ v8f wmma_bf16(const ABFrag& a, const ABFrag& b, v8f c) {
    return __builtin_amdgcn_wmma_f32_16x16x32_bf16(false, a.bf, false, b.bf,
                                                   (short)0, c, false, false);
}

__device__ __forceinline__ unsigned rotl32(unsigned v, int s) {
    return (v << s) | (v >> (32 - s));
}

// Threefry-2x32, 20 rounds (matches JAX threefry2x32)
__device__ __forceinline__ void threefry2x32(unsigned k0, unsigned k1,
                                             unsigned x0, unsigned x1,
                                             unsigned& y0, unsigned& y1) {
    const int R[8] = {13, 15, 26, 6, 17, 29, 16, 24};
    unsigned ks[3] = {k0, k1, k0 ^ k1 ^ 0x1BD11BDAu};
    x0 += ks[0]; x1 += ks[1];
#pragma unroll
    for (int i = 0; i < 5; ++i) {
        const int base = (i & 1) ? 4 : 0;
#pragma unroll
        for (int r = 0; r < 4; ++r) { x0 += x1; x1 = rotl32(x1, R[base + r]); x1 ^= x0; }
        x0 += ks[(i + 1) % 3];
        x1 += ks[(i + 2) % 3] + (unsigned)(i + 1);
    }
    y0 = x0; y1 = x1;
}

// ---------------- one-time weight conversion (f32 -> bf16) ----------------
__global__ void convert_weights_kernel(const float* __restrict__ linW,
                                       const float* __restrict__ whh,
                                       const float* __restrict__ wih,
                                       unsigned short* __restrict__ linWb,
                                       unsigned short* __restrict__ whhb,
                                       unsigned short* __restrict__ wihb) {
    const int N1 = VOCAB * HDIM, N2 = 3 * HDIM * HDIM, N3 = 3 * HDIM * EDIM;
    const int stride = gridDim.x * blockDim.x;
    for (int i = blockIdx.x * blockDim.x + threadIdx.x; i < N1; i += stride)
        linWb[i] = (unsigned short)f32_to_bf16_bits(linW[i]);
    for (int i = blockIdx.x * blockDim.x + threadIdx.x; i < N2; i += stride)
        whhb[i] = (unsigned short)f32_to_bf16_bits(whh[i]);
    for (int i = blockIdx.x * blockDim.x + threadIdx.x; i < N3; i += stride)
        wihb[i] = (unsigned short)f32_to_bf16_bits(wih[i]);
}

// ---------------- init: h0 = concat(init_hidden, att_emb); emb0 gather ----------
__global__ void init_state_kernel(const int* __restrict__ inputs,
                                  const float* __restrict__ init_hidden,
                                  const float* __restrict__ att_emb,
                                  const float* __restrict__ emb_table,
                                  float* __restrict__ h_f,
                                  unsigned short* __restrict__ h_b,
                                  unsigned short* __restrict__ emb_b) {
    const int stride = gridDim.x * blockDim.x;
    for (int i = blockIdx.x * blockDim.x + threadIdx.x; i < BATCH * HDIM; i += stride) {
        const int m = i >> 10, k = i & (HDIM - 1);
        const float v = (k < 768) ? init_hidden[m * 768 + k] : att_emb[m * EDIM + (k - 768)];
        h_f[i] = v;
        h_b[i] = (unsigned short)f32_to_bf16_bits(v);
    }
    for (int i = blockIdx.x * blockDim.x + threadIdx.x; i < BATCH * EDIM; i += stride) {
        const int m = i >> 8, k = i & (EDIM - 1);
        const int sid = inputs[m];  // inputs[:,0]
        emb_b[i] = (unsigned short)f32_to_bf16_bits(emb_table[(size_t)sid * EDIM + k]);
    }
}

// ---------------- GRU step: 64 blocks x 256 thr; 8 waves = 2 rowTiles x 4 parts -----
// parts: 0 = r-sum (gi+gh), 1 = z-sum (gi+gh), 2 = i_n (gi only), 3 = h_n (gh only)
__global__ void gru_step_kernel(const float* __restrict__ hin_f,
                                const unsigned short* __restrict__ hin_b,
                                const unsigned short* __restrict__ emb_b,
                                const unsigned short* __restrict__ wih,
                                const unsigned short* __restrict__ whh,
                                const float* __restrict__ bih,
                                const float* __restrict__ bhh,
                                float* __restrict__ hout_f,
                                unsigned short* __restrict__ hout_b) {
    __shared__ float lds[4][BATCH][16];
    const int tid = threadIdx.x, lane = tid & 31, w = tid >> 5;
    const int rt = w & 1, part = w >> 1;
    const int j0 = blockIdx.x * 16;

    v8f acc = {};
    ABFrag a, b;
    if (part <= 2) {  // input-gate contribution (K = 256)
        const int ncol = part * HDIM + j0;
#pragma unroll
        for (int k0 = 0; k0 < EDIM; k0 += 32) {
            load_a(a, emb_b, rt * 16, EDIM, k0, lane);
            load_b(b, wih, ncol, EDIM, k0, lane);
            acc = wmma_bf16(a, b, acc);
        }
    }
    if (part != 2) {  // hidden-gate contribution (K = 1024)
        const int g = (part == 3) ? 2 : part;
        const int ncol = g * HDIM + j0;
        for (int k0 = 0; k0 < HDIM; k0 += 32) {
            load_a(a, hin_b, rt * 16, HDIM, k0, lane);
            load_b(b, whh, ncol, HDIM, k0, lane);
            acc = wmma_bf16(a, b, acc);
        }
    }
    {   // C layout: N = lane%16, M = vgpr + 8*(lane/16)
        const int n = lane & 15, rbase = rt * 16 + (lane >> 4) * 8;
#pragma unroll
        for (int v = 0; v < 8; ++v) lds[part][rbase + v][n] = acc[v];
    }
    __syncthreads();

    for (int idx = tid; idx < BATCH * 16; idx += 256) {
        const int m = idx >> 4, c = idx & 15, j = j0 + c;
        const float rs = lds[0][m][c] + bih[j] + bhh[j];
        const float zs = lds[1][m][c] + bih[HDIM + j] + bhh[HDIM + j];
        const float iv = lds[2][m][c] + bih[2 * HDIM + j];
        const float hv = lds[3][m][c] + bhh[2 * HDIM + j];
        const float r = 1.f / (1.f + __expf(-rs));
        const float z = 1.f / (1.f + __expf(-zs));
        const float n = tanhf(iv + r * hv);
        const float hold = hin_f[m * HDIM + j];
        const float hnew = (1.f - z) * n + z * hold;
        hout_f[m * HDIM + j] = hnew;
        hout_b[m * HDIM + j] = (unsigned short)f32_to_bf16_bits(hnew);
    }
}

// ---------------- logits: 125 blocks x 256 thr -----------------------------------
// Each wave owns 4 consecutive 16-col tiles (64 vocab cols) x one 16-row tile.
// Per k-step: 1 A-load shared across 4 WMMAs (A reuse), 4 B-loads (lin_W stream).
__global__ void logits_kernel(const unsigned short* __restrict__ hb,
                              const unsigned short* __restrict__ linWb,
                              const float* __restrict__ linb,
                              float* __restrict__ out, int t) {
    const int tid = threadIdx.x, lane = tid & 31, w = tid >> 5;
    const int rt = w & 1;
    const int group = blockIdx.x * 4 + (w >> 1);  // 500 column groups of 64 cols
    const int v0 = group * 64;

    v8f acc0 = {}, acc1 = {}, acc2 = {}, acc3 = {};
    ABFrag a, b0, b1, b2, b3;
    for (int k0 = 0; k0 < HDIM; k0 += 32) {
        load_a(a, hb, rt * 16, HDIM, k0, lane);
        load_b(b0, linWb, v0,      HDIM, k0, lane);
        load_b(b1, linWb, v0 + 16, HDIM, k0, lane);
        load_b(b2, linWb, v0 + 32, HDIM, k0, lane);
        load_b(b3, linWb, v0 + 48, HDIM, k0, lane);
        acc0 = wmma_bf16(a, b0, acc0);
        acc1 = wmma_bf16(a, b1, acc1);
        acc2 = wmma_bf16(a, b2, acc2);
        acc3 = wmma_bf16(a, b3, acc3);
    }

    const int n = lane & 15;
    const int rbase = rt * 16 + (lane >> 4) * 8;
    const v8f accs[4] = {acc0, acc1, acc2, acc3};
#pragma unroll
    for (int c = 0; c < 4; ++c) {
        const int col = v0 + c * 16 + n;
        const float bias = linb[col];
#pragma unroll
        for (int v = 0; v < 8; ++v) {
            const int m = rbase + v;  // batch row
            out[(size_t)(m * TSTEP + t) * VOCAB + col] = accs[c][v] + bias;
        }
    }
}

// ---------------- sample: gumbel-argmax with JAX-style threefry, then emb gather ----
__global__ void sample_kernel(const float* __restrict__ logits_base,
                              const float* __restrict__ emb_table,
                              unsigned short* __restrict__ emb_b,
                              float* __restrict__ sid_out, int t) {
    __shared__ float smax[256];
    __shared__ int   sidx[256];
    __shared__ int   swin;
    const int m = blockIdx.x, tid = threadIdx.x;

    // key = fold_in(key(42), t) = threefry2x32((0,42), (0,t))
    unsigned fk0, fk1;
    threefry2x32(0u, 42u, 0u, (unsigned)t, fk0, fk1);

    const float* lrow = logits_base + (size_t)(m * TSTEP + t) * VOCAB;
    const unsigned HALFN = (unsigned)(BATCH * VOCAB / 2);

    float best = -3.0e38f; int bi = 0;
    for (int v = tid; v < VOCAB; v += 256) {
        const unsigned idx = (unsigned)(m * VOCAB + v);
        unsigned c0, c1, y0, y1;
        if (idx < HALFN) { c0 = idx; c1 = idx + HALFN; }
        else             { c0 = idx - HALFN; c1 = idx; }
        threefry2x32(fk0, fk1, c0, c1, y0, y1);
        const unsigned bits = (idx < HALFN) ? y0 : y1;
        const float u = __uint_as_float((bits >> 9) | 0x3f800000u) - 1.0f;
        const float g = -logf(-log1pf(-u));
        const float s = lrow[v] + g;
        if (s > best) { best = s; bi = v; }
    }
    smax[tid] = best; sidx[tid] = bi;
    __syncthreads();
    for (int off = 128; off > 0; off >>= 1) {
        if (tid < off && smax[tid + off] > smax[tid]) {
            smax[tid] = smax[tid + off]; sidx[tid] = sidx[tid + off];
        }
        __syncthreads();
    }
    if (tid == 0) {
        swin = sidx[0];
        sid_out[m * TSTEP + t] = (float)sidx[0];
    }
    __syncthreads();
    const int sid = swin;
    emb_b[m * EDIM + tid] = (unsigned short)f32_to_bf16_bits(
        emb_table[(size_t)sid * EDIM + tid]);
}

// ---------------- launch ----------------
extern "C" void kernel_launch(void* const* d_in, const int* in_sizes, int n_in,
                              void* d_out, int out_size, void* d_ws, size_t ws_size,
                              hipStream_t stream) {
    const int*   inputs      = (const int*)d_in[0];
    const float* init_hidden = (const float*)d_in[2];
    const float* att_emb     = (const float*)d_in[3];
    const float* emb_table   = (const float*)d_in[4];
    const float* w_ih        = (const float*)d_in[5];
    const float* w_hh        = (const float*)d_in[6];
    const float* b_ih        = (const float*)d_in[7];
    const float* b_hh        = (const float*)d_in[8];
    const float* lin_W       = (const float*)d_in[9];
    const float* lin_b       = (const float*)d_in[10];

    float* out     = (float*)d_out;                        // logits, then ids
    float* sid_out = out + (size_t)BATCH * TSTEP * VOCAB;  // 32x32 ids as float

    // workspace carve-up (256B aligned)
    char* ws = (char*)d_ws;
    size_t cur = 0;
    auto carve = [&](size_t bytes) { size_t o = cur; cur = (cur + bytes + 255) & ~(size_t)255; return o; };
    unsigned short* linWb = (unsigned short*)(ws + carve((size_t)VOCAB * HDIM * 2));
    unsigned short* whhb  = (unsigned short*)(ws + carve((size_t)3 * HDIM * HDIM * 2));
    unsigned short* wihb  = (unsigned short*)(ws + carve((size_t)3 * HDIM * EDIM * 2));
    float*          hf[2];
    unsigned short* hb[2];
    hf[0] = (float*)(ws + carve((size_t)BATCH * HDIM * 4));
    hf[1] = (float*)(ws + carve((size_t)BATCH * HDIM * 4));
    hb[0] = (unsigned short*)(ws + carve((size_t)BATCH * HDIM * 2));
    hb[1] = (unsigned short*)(ws + carve((size_t)BATCH * HDIM * 2));
    unsigned short* embb = (unsigned short*)(ws + carve((size_t)BATCH * EDIM * 2));

    convert_weights_kernel<<<2048, 256, 0, stream>>>(lin_W, w_hh, w_ih, linWb, whhb, wihb);
    init_state_kernel<<<128, 256, 0, stream>>>(inputs, init_hidden, att_emb, emb_table,
                                               hf[0], hb[0], embb);

    for (int t = 0; t < TSTEP; ++t) {
        const int src = t & 1, dst = (t + 1) & 1;
        gru_step_kernel<<<HDIM / 16, 256, 0, stream>>>(
            hf[src], hb[src], embb, wihb, whhb, b_ih, b_hh, hf[dst], hb[dst]);
        logits_kernel<<<VOCAB / 16 / 16, 256, 0, stream>>>(hb[dst], linWb, lin_b, out, t);
        sample_kernel<<<BATCH, 256, 0, stream>>>(out, emb_table, embb, sid_out, t);
    }
}
